// SplineCouplingLayer_85040352461492
// MI455X (gfx1250) — compile-verified, matrix operand order checked by reading, not run on hardware
//
#include <hip/hip_runtime.h>
#include <hip/hip_bf16.h>

typedef _Float16 h8  __attribute__((ext_vector_type(8)));
typedef _Float16 h16 __attribute__((ext_vector_type(16)));
typedef float    f8  __attribute__((ext_vector_type(8)));

#define B_ROWS 131072
#define DIN    32
#define MASKD  16
#define CTX    64
#define HID    256
#define NBINS  16
#define TDIM   16            // DIN - MASKD
#define PCOL   49            // 3*NBINS + 1
#define NOUT   (TDIM*PCOL)   // 784

#define TM      64           // rows per workgroup
#define THREADS 512          // 16 wave32s
#define NWAVES  (THREADS/32)

#define KPAD1 96             // 80 padded to multiple of 32
// LDS row strides (in halfs / floats), padded for bank-conflict-free b128/b32
#define SA 104               // f16, 208B rows (13*16B)
#define SH 264               // f16, 528B rows (33*16B)
#define SP 788               // f32

// LDS layout (bytes)
#define OFF_A   0                          // 64*104*2 = 13312
#define OFF_H1  13312                      // 64*264*2 = 33792
#define OFF_H2  47104                      // 33792
#define OFF_P   80896                      // 64*788*4 = 201728
#define OFF_LAD 282624                     // 64*4
#define SMEM_BYTES 282880                  // < 320KB WGP LDS

// ---------------------------------------------------------------------------
// Weight conversion: f32 row-major (K,N) -> f16 N-major (N,K) with K padding.
// Tiny (291K elements); output stays resident in the 192MB L2 for all blocks.
// ---------------------------------------------------------------------------
__global__ void cvt_weights(const float* __restrict__ W1,
                            const float* __restrict__ W2,
                            const float* __restrict__ W3,
                            _Float16* __restrict__ wts) {
  const int n1 = HID * KPAD1;        // W1t: 256 x 96
  const int n2 = HID * HID;          // W2t: 256 x 256
  const int n3 = NOUT * HID;         // W3t: 784 x 256
  int i = blockIdx.x * blockDim.x + threadIdx.x;
  if (i < n1) {
    int n = i / KPAD1, k = i - n * KPAD1;
    wts[i] = (k < MASKD + CTX) ? (_Float16)W1[k * HID + n] : (_Float16)0.f;
  } else if (i < n1 + n2) {
    int j = i - n1;
    int n = j / HID, k = j - n * HID;
    wts[i] = (_Float16)W2[k * HID + n];
  } else if (i < n1 + n2 + n3) {
    int j = i - n1 - n2;
    int n = j / HID, k = j - n * HID;
    wts[i] = (_Float16)W3[k * NOUT + n];
  }
}

// ---------------------------------------------------------------------------
// WMMA fragment loader. 16-bit A-matrix 16x32 layout (ISA 7.12.2):
//   lanes 0-15 / 16-31 hold M=0-15; v0-3: K=(0..7 | 8..15), v4-7: +16.
// Each lane: two 16B contiguous loads (ds_load_b128 / global_load_b128).
// B (32x16) is stored N-major so the same loader applies with row == N.
// ---------------------------------------------------------------------------
__device__ __forceinline__ h16 load_frag(const _Float16* tile, int stride, int lane) {
  const int r  = lane & 15;
  const int kb = (lane >> 4) << 3;     // 0 or 8
  const _Float16* q = tile + r * stride + kb;
  h8 lo = *(const h8*)(q);
  h8 hi = *(const h8*)(q + 16);
  return __builtin_shufflevector(lo, hi, 0,1,2,3,4,5,6,7,8,9,10,11,12,13,14,15);
}

__device__ __forceinline__ f8 wmma_f16(h16 a, h16 b, f8 c) {
  return __builtin_amdgcn_wmma_f32_16x16x32_f16(false, a, false, b, (short)0, c, false, false);
}

__device__ __forceinline__ float softplusf(float v) {
  return fmaxf(v, 0.f) + log1pf(expf(-fabsf(v)));
}

// ---------------------------------------------------------------------------
// Fused coupling layer: 3 WMMA GEMMs (f16 in, f32 acc) + RQ-spline, 64 rows/WG
// Wave tiling: wave = wq*4 + wr; wq -> N-tile phase, wr... see below:
//   row-tile  mt = wave & 3           (4 row tiles of 16)
//   N stride: nt = (wave >> 2) + 4*j  (4 waves per row tile)
// A fragments preloaded once per wave (full K in registers), only B streams.
// ---------------------------------------------------------------------------
__global__ void __launch_bounds__(THREADS)
spline_coupling_kernel(const float* __restrict__ inputs,
                       const float* __restrict__ context,
                       const float* __restrict__ b1,
                       const float* __restrict__ b2,
                       const float* __restrict__ b3,
                       const _Float16* __restrict__ wts,
                       float* __restrict__ out,
                       float* __restrict__ lad_out) {
  extern __shared__ __align__(16) char smem[];
  _Float16* Ain = (_Float16*)(smem + OFF_A);
  _Float16* H1  = (_Float16*)(smem + OFF_H1);
  _Float16* H2  = (_Float16*)(smem + OFF_H2);
  float*    Ps  = (float*)(smem + OFF_P);
  float*    Lad = (float*)(smem + OFF_LAD);

  const _Float16* W1t = wts;                              // 256 x 96
  const _Float16* W2t = wts + HID * KPAD1;                // 256 x 256
  const _Float16* W3t = wts + HID * KPAD1 + HID * HID;    // 784 x 256

  const int  tid  = threadIdx.x;
  const int  lane = tid & 31;
  // force wave id into an SGPR: scalar tile loops, uniform EXEC around WMMA
  const int  wave = __builtin_amdgcn_readfirstlane(tid >> 5);
  const int  mt   = wave & 3;        // row tile (rows mt*16 .. mt*16+15)
  const int  wr   = wave >> 2;       // column-tile phase (0..3)
  const long rowBase = (long)blockIdx.x * TM;

  // per-lane C-tile coordinates (ISA 7.12.2 f32 C/D layout)
  const int cn  = lane & 15;
  const int cm0 = (lane >> 4) << 3;

  // ---- Stage net_in = [masked | context | 0-pad] as f16; passthrough copy ----
  for (int e = tid; e < TM * KPAD1; e += THREADS) {
    int r = e / KPAD1, c = e - r * KPAD1;
    long g = rowBase + r;
    float v = (c < MASKD)        ? inputs[g * DIN + c]
            : (c < MASKD + CTX)  ? context[g * CTX + (c - MASKD)]
                                 : 0.f;
    Ain[r * SA + c] = (_Float16)v;
  }
  for (int e = tid; e < TM * MASKD; e += THREADS) {
    int r = e >> 4, c = e & 15;
    long g = rowBase + r;
    out[g * DIN + c] = inputs[g * DIN + c];
  }
  if (tid < TM) Lad[tid] = 0.f;
  __syncthreads();

  // ---- GEMM1: (64x96) @ (96x256) -> H1, fused bias + ReLU, f16 ----
  {
    h16 afrag[KPAD1 / 32];
#pragma unroll
    for (int kt = 0; kt < KPAD1 / 32; ++kt)
      afrag[kt] = load_frag(Ain + (mt * 16) * SA + kt * 32, SA, lane);
    for (int nt = wr; nt < HID / 16; nt += 4) {
      f8 acc = {0.f, 0.f, 0.f, 0.f, 0.f, 0.f, 0.f, 0.f};
#pragma unroll
      for (int kt = 0; kt < KPAD1 / 32; ++kt) {
        h16 b = load_frag(W1t + (nt * 16) * KPAD1 + kt * 32, KPAD1, lane);
        acc = wmma_f16(afrag[kt], b, acc);
      }
      float bias = b1[nt * 16 + cn];
#pragma unroll
      for (int r = 0; r < 8; ++r) {
        float v = fmaxf(acc[r] + bias, 0.f);
        H1[(mt * 16 + cm0 + r) * SH + nt * 16 + cn] = (_Float16)v;
      }
    }
  }
  __syncthreads();

  // ---- GEMM2: (64x256) @ (256x256) -> H2, fused bias + ReLU, f16 ----
  {
    h16 afrag[HID / 32];
#pragma unroll
    for (int kt = 0; kt < HID / 32; ++kt)
      afrag[kt] = load_frag(H1 + (mt * 16) * SH + kt * 32, SH, lane);
    for (int nt = wr; nt < HID / 16; nt += 4) {
      f8 acc = {0.f, 0.f, 0.f, 0.f, 0.f, 0.f, 0.f, 0.f};
#pragma unroll
      for (int kt = 0; kt < HID / 32; ++kt) {
        h16 b = load_frag(W2t + (nt * 16) * HID + kt * 32, HID, lane);
        acc = wmma_f16(afrag[kt], b, acc);
      }
      float bias = b2[nt * 16 + cn];
#pragma unroll
      for (int r = 0; r < 8; ++r) {
        float v = fmaxf(acc[r] + bias, 0.f);
        H2[(mt * 16 + cm0 + r) * SH + nt * 16 + cn] = (_Float16)v;
      }
    }
  }
  __syncthreads();

  // ---- GEMM3: (64x256) @ (256x784) -> Ps (f32 + bias) in LDS ----
  {
    h16 afrag[HID / 32];
#pragma unroll
    for (int kt = 0; kt < HID / 32; ++kt)
      afrag[kt] = load_frag(H2 + (mt * 16) * SH + kt * 32, SH, lane);
    for (int nt = wr; nt < PCOL; nt += 4) {
      f8 acc = {0.f, 0.f, 0.f, 0.f, 0.f, 0.f, 0.f, 0.f};
#pragma unroll
      for (int kt = 0; kt < HID / 32; ++kt) {
        h16 b = load_frag(W3t + (nt * 16) * HID + kt * 32, HID, lane);
        acc = wmma_f16(afrag[kt], b, acc);
      }
      float bias = b3[nt * 16 + cn];
#pragma unroll
      for (int r = 0; r < 8; ++r)
        Ps[(mt * 16 + cm0 + r) * SP + nt * 16 + cn] = acc[r] + bias;
    }
  }
  __syncthreads();

  // ---- RQ spline: one (row, dim) pair per iteration ----
  for (int p = tid; p < TM * TDIM; p += THREADS) {
    int  r   = p >> 4;
    int  dim = p & 15;
    long g   = rowBase + r;
    float x  = inputs[g * DIN + MASKD + dim];
    bool inside = (x >= -3.f) && (x <= 3.f);
    float xc = fminf(fmaxf(x, -3.f), 3.f);
    const float* P = Ps + r * SP + dim * PCOL;

    // stable softmax normalizers for widths / heights
    float wmax = P[0], hmax = P[NBINS];
#pragma unroll
    for (int k = 1; k < NBINS; ++k) {
      wmax = fmaxf(wmax, P[k]);
      hmax = fmaxf(hmax, P[NBINS + k]);
    }
    float wsum = 0.f, hsum = 0.f;
#pragma unroll
    for (int k = 0; k < NBINS; ++k) {
      wsum += expf(P[k] - wmax);
      hsum += expf(P[NBINS + k] - hmax);
    }
    const float cws = (1.f - 0.001f * NBINS) / wsum;
    const float chs = (1.f - 0.001f * NBINS) / hsum;

    // fused cumsum + bin search
    float leftX = -3.f, leftY = -3.f;
    float icw = -3.f, ibw = 1.f, ich = -3.f, ibh = 1.f;
    int idx = -1;
#pragma unroll
    for (int k = 0; k < NBINS; ++k) {
      float wk = 6.f * (0.001f + cws * expf(P[k] - wmax));
      float hk = 6.f * (0.001f + chs * expf(P[NBINS + k] - hmax));
      bool pick = (idx < 0) && ((xc < leftX + wk) || (k == NBINS - 1));
      if (pick) { idx = k; icw = leftX; ibw = wk; ich = leftY; ibh = hk; }
      leftX += wk;
      leftY += hk;
    }

    float d0 = 0.001f + softplusf(P[2 * NBINS + idx]);
    float d1 = 0.001f + softplusf(P[2 * NBINS + idx + 1]);
    float delta = ibh / ibw;
    float theta = (xc - icw) / ibw;
    float omt   = 1.f - theta;
    float t1m   = theta * omt;
    float num   = ibh * (delta * theta * theta + d0 * t1m);
    float den   = delta + (d0 + d1 - 2.f * delta) * t1m;
    float y     = ich + num / den;
    float dnum  = delta * delta * (d1 * theta * theta + 2.f * delta * t1m + d0 * omt * omt);
    float lad   = logf(dnum) - 2.f * logf(den);
    y   = inside ? y : x;
    lad = inside ? lad : 0.f;
    out[g * DIN + MASKD + dim] = y;
    atomicAdd(&Lad[r], lad);
  }
  __syncthreads();

  if (tid < TM) lad_out[rowBase + tid] = Lad[tid];
}

// ---------------------------------------------------------------------------
extern "C" void kernel_launch(void* const* d_in, const int* in_sizes, int n_in,
                              void* d_out, int out_size, void* d_ws, size_t ws_size,
                              hipStream_t stream) {
  const float* inputs  = (const float*)d_in[0];
  const float* context = (const float*)d_in[1];
  const float* W1      = (const float*)d_in[2];
  const float* b1      = (const float*)d_in[3];
  const float* W2      = (const float*)d_in[4];
  const float* b2      = (const float*)d_in[5];
  const float* W3      = (const float*)d_in[6];
  const float* b3      = (const float*)d_in[7];

  float* out = (float*)d_out;
  float* lad = out + (long)B_ROWS * DIN;
  _Float16* wts = (_Float16*)d_ws;

  const int ncvt = HID * KPAD1 + HID * HID + NOUT * HID;   // 290,816 f16
  hipLaunchKernelGGL(cvt_weights, dim3((ncvt + 255) / 256), dim3(256), 0, stream,
                     W1, W2, W3, wts);

  hipLaunchKernelGGL(spline_coupling_kernel, dim3(B_ROWS / TM), dim3(THREADS),
                     SMEM_BYTES, stream, inputs, context, b1, b2, b3, wts, out, lad);
}
